// MultiScaleAttention_18468359372968
// MI455X (gfx1250) — matrix operand order repaired and checked by reading, not compiled, not run on hardware
//
#include <hip/hip_runtime.h>
#include <hip/hip_bf16.h>
#include <cstdint>
#include <cstddef>

// ---------------------------------------------------------------------------
// MultiScaleAttention for MI455X (gfx1250, wave32, WMMA).
// Pipeline: fp32->bf16 convert -> 3x bf16 WMMA GEMM (Q,K,V, double-buffered
// async-to-LDS staging) -> windowed flash-attention (bf16 WMMA, fp32 softmax)
// into parity slot buffers -> combine/average -> bf16 WMMA GEMM with Wo.
// Workspace requirement: ~310 MB (see kernel_launch layout).
// ---------------------------------------------------------------------------

typedef __bf16 bf16_t;
typedef __attribute__((ext_vector_type(16))) bf16_t v16bf;
typedef __attribute__((ext_vector_type(8)))  float  v8f;
typedef __attribute__((ext_vector_type(4)))  int    v4i;

#define DMODEL 1024
#define NHEADS 16
#define HDIM   64
#define NWIN   31
#define BATCH  4
#define SEQ    4096

// ---- gfx1250 async global->LDS copy (ASYNCcnt) with safe fallbacks --------
#if defined(__has_builtin)
#  if __has_builtin(__builtin_amdgcn_global_load_async_to_lds_b128)
#    define HAVE_ASYNC_LDS 1
#  else
#    define HAVE_ASYNC_LDS 0
#  endif
#  if __has_builtin(__builtin_amdgcn_s_wait_asynccnt)
#    define WAIT_ASYNC(n) __builtin_amdgcn_s_wait_asynccnt(n)
#  else
#    define WAIT_ASYNC(n) asm volatile("s_wait_asynccnt %0" ::"n"(n) : "memory")
#  endif
#else
#  define HAVE_ASYNC_LDS 0
#  define WAIT_ASYNC(n)
#endif

// builtin signature: (v4i AS1* global_src, v4i AS3* lds_dst, imm off, imm cpol)
typedef __attribute__((address_space(1))) v4i* gv4i_p;
typedef __attribute__((address_space(3))) v4i* lv4i_p;

__device__ __forceinline__ void async_copy_b128(const bf16_t* g, bf16_t* l) {
#if HAVE_ASYNC_LDS
  __builtin_amdgcn_global_load_async_to_lds_b128((gv4i_p)(void*)g,
                                                 (lv4i_p)(void*)l, 0, 0);
#else
  *(uint4*)l = *(const uint4*)g;
#endif
}

// ---- WMMA fragment helpers (per CDNA5 ISA §7.12.2 layouts, wave32) --------
// A (16x32 bf16): lane L holds row L&15; halves 0..7 -> K = (L>>4)*8 + 0..7,
// halves 8..15 -> K = 16 + (L>>4)*8 + 0..7. Source must be row-major.
__device__ __forceinline__ v16bf load_a_frag(const bf16_t* base, int lda) {
  const int lane = threadIdx.x & 31;
  const int row  = lane & 15;
  const int kh   = lane >> 4;
  const bf16_t* p = base + row * lda + kh * 8;
  union { uint4 u[2]; v16bf v; } r;
  r.u[0] = *(const uint4*)(p);        // K = kh*8 + 0..7
  r.u[1] = *(const uint4*)(p + 16);   // K = 16 + kh*8 + 0..7
  return r.v;
}

// B (32x16 bf16): lane L holds column L&15, K = (L>>4)*8 + 0..7, 16 + (L>>4)*8 + 0..7.
// Source must be B^T row-major (column of B contiguous).
__device__ __forceinline__ v16bf load_b_frag(const bf16_t* baseT, int ldt) {
  const int lane = threadIdx.x & 31;
  const int col  = lane & 15;
  const int klo  = (lane >> 4) * 8;
  const bf16_t* p = baseT + col * ldt + klo;
  union { uint4 u[2]; v16bf v; } r;
  r.u[0] = *(const uint4*)(p);
  r.u[1] = *(const uint4*)(p + 16);
  return r.v;
}

__device__ __forceinline__ v8f wmma_bf16(v16bf a, v16bf b, v8f c) {
  return __builtin_amdgcn_wmma_f32_16x16x32_bf16(false, a, false, b,
                                                 (short)0, c, false, false);
}

// ---- Elementwise kernels ---------------------------------------------------
__global__ void convert_f32_to_bf16(const float* __restrict__ in,
                                    bf16_t* __restrict__ out, size_t n) {
  size_t i = (size_t)blockIdx.x * blockDim.x + threadIdx.x;
  const size_t stride = (size_t)gridDim.x * blockDim.x;
  for (; i < n; i += stride) out[i] = (bf16_t)in[i];
}

__global__ void zero_f32(float* __restrict__ p, size_t n) {
  size_t i = (size_t)blockIdx.x * blockDim.x + threadIdx.x;
  const size_t stride = (size_t)gridDim.x * blockDim.x;
  for (; i < n; i += stride) p[i] = 0.0f;
}

// avg of the two parity slots by per-token window count, emit bf16 for Wo GEMM
__global__ void combine_slots(const float* __restrict__ s0,
                              const float* __restrict__ s1,
                              bf16_t* __restrict__ out, size_t n) {
  size_t i = (size_t)blockIdx.x * blockDim.x + threadIdx.x;
  const size_t stride = (size_t)gridDim.x * blockDim.x;
  for (; i < n; i += stride) {
    const int token = (int)((i / DMODEL) % SEQ);
    const float inv = (token >= 128 && token < SEQ - 128) ? 0.5f : 1.0f;
    out[i] = (bf16_t)((s0[i] + s1[i]) * inv);
  }
}

// ---- bf16 WMMA GEMM: C[MxN] = A[MxK] * B[KxN] + bias, fp32 accumulate -----
// Block tile 256x64, K-step 32, double-buffered LDS with async-to-LDS A
// staging. 256 threads = 8 waves; wave w -> rows [w*32, w*32+32) as two
// m-tiles sharing the 4 B-fragments (8 WMMAs per 12 ds_load_b128).
template <bool OUT_F32>
__global__ __launch_bounds__(256)
void gemm_bias(const bf16_t* __restrict__ A, const bf16_t* __restrict__ Bw,
               const float* __restrict__ bias, void* __restrict__ Cout,
               int M, int N, int K) {
  __shared__ bf16_t As[2][256][40];   // row-major A tiles (+pad)
  __shared__ bf16_t Bt[2][64][40];    // transposed B tiles: [n][k] (+pad)

  const int tid  = threadIdx.x;
  const int wave = tid >> 5;
  const int lane = tid & 31;
  const int mblk = blockIdx.x * 256;
  const int nblk = blockIdx.y * 64;

  v8f acc[2][4];
#pragma unroll
  for (int mt = 0; mt < 2; ++mt)
#pragma unroll
    for (int t = 0; t < 4; ++t) acc[mt][t] = v8f{};

  const int NK = K / 32;

  // A: thread t owns tile row t -> 4 async b128 copies (64 B contiguous).
  auto stageA = [&](int buf, int kt) {
    const bf16_t* src = A + (size_t)(mblk + tid) * K + kt;
#pragma unroll
    for (int c = 0; c < 4; ++c)
      async_copy_b128(src + c * 8, &As[buf][tid][c * 8]);
  };
  // B: 32x64 halves with transpose into [n][k]; 8 halves per thread.
  auto stageB = [&](int buf, int kt) {
    const int k  = tid >> 3;
    const int n0 = (tid & 7) * 8;
    const bf16_t* src = Bw + (size_t)(kt + k) * N + nblk + n0;
    __builtin_prefetch(src + 32, 0, 0);
    uint4 vv = *(const uint4*)src;
    bf16_t tmp[8];
    *(uint4*)tmp = vv;
#pragma unroll
    for (int i = 0; i < 8; ++i) Bt[buf][n0 + i][k] = tmp[i];
  };

  stageA(0, 0);
  stageB(0, 0);

  for (int n = 0; n < NK; ++n) {
    const int buf = n & 1;
    if (n + 1 < NK) {                 // prefetch next tile into other buffer
      stageA(buf ^ 1, (n + 1) * 32);
      stageB(buf ^ 1, (n + 1) * 32);
    }
#if HAVE_ASYNC_LDS
    // async loads complete in order: <=4 outstanding proves tile n landed
    if (n + 1 < NK) WAIT_ASYNC(4); else WAIT_ASYNC(0);
#endif
    __syncthreads();

    const v16bf a0 = load_a_frag(&As[buf][wave * 32][0], 40);
    const v16bf a1 = load_a_frag(&As[buf][wave * 32 + 16][0], 40);
#pragma unroll
    for (int t = 0; t < 4; ++t) {
      const v16bf b = load_b_frag(&Bt[buf][t * 16][0], 40);
      acc[0][t] = wmma_bf16(a0, b, acc[0][t]);
      acc[1][t] = wmma_bf16(a1, b, acc[1][t]);
    }
    __syncthreads();                  // tile n consumed; safe to overwrite
  }

  // epilogue: C layout VGPR r / lanes -> (row = r + 8*(lane>>4), col = lane&15)
  const int lhalf = lane >> 4;
  const int lcol  = lane & 15;
#pragma unroll
  for (int mt = 0; mt < 2; ++mt) {
#pragma unroll
    for (int t = 0; t < 4; ++t) {
      const int colg = nblk + t * 16 + lcol;
      const float bb = bias[colg];
#pragma unroll
      for (int r = 0; r < 8; ++r) {
        const int rowg = mblk + wave * 32 + mt * 16 + r + 8 * lhalf;
        const float val = acc[mt][t][r] + bb;
        if (OUT_F32)
          ((float*)Cout)[(size_t)rowg * N + colg] = val;
        else
          ((bf16_t*)Cout)[(size_t)rowg * N + colg] = (bf16_t)val;
      }
    }
  }
}

// ---- Windowed attention: one block per (window, head, batch) --------------
// 512 threads = 16 waves; wave w owns window rows [w*16, w*16+16).
// Flash-style online softmax over 4 chunks of 64 key tokens.
__global__ __launch_bounds__(512)
void attn_kernel(const bf16_t* __restrict__ Q, const bf16_t* __restrict__ K,
                 const bf16_t* __restrict__ V, float* __restrict__ slot0,
                 float* __restrict__ slot1) {
  __shared__ bf16_t Ks[64][72];       // K chunk, row-major [token][feat] = B^T
  __shared__ bf16_t Vt[64][72];       // V chunk transposed [feat][token] = B^T
  __shared__ bf16_t Ps[16][16 * 72];  // per-wave probability tile (A layout)

  const int w = blockIdx.x;           // window (start = 128*w)
  const int h = blockIdx.y;           // head
  const int b = blockIdx.z;           // batch
  const int wstart = w * 128;

  const int tid   = threadIdx.x;
  const int wave  = tid >> 5;
  const int lane  = tid & 31;
  const int lhalf = lane >> 4;
  const int lcol  = lane & 15;

  const size_t bh_off = (size_t)b * SEQ * DMODEL + (size_t)h * HDIM;
  const bf16_t* Qg = Q + bh_off;
  const bf16_t* Kg = K + bh_off;
  const bf16_t* Vg = V + bh_off;

  // Q fragments for this wave's 16 rows (constant across chunks)
  const bf16_t* qbase = Qg + (size_t)(wstart + wave * 16) * DMODEL;
  const v16bf aq0 = load_a_frag(qbase, DMODEL);
  const v16bf aq1 = load_a_frag(qbase + 32, DMODEL);

  v8f o0 = {}, o1 = {}, o2 = {}, o3 = {};
  float rm[8], rl[8];
#pragma unroll
  for (int r = 0; r < 8; ++r) { rm[r] = -1e30f; rl[r] = 0.0f; }

  const float scale = 0.125f;  // 1/sqrt(64)
  bf16_t* Pw = &Ps[wave][0];

  for (int c = 0; c < 4; ++c) {
    __syncthreads();
    { // stage K (async->LDS) and V^T chunk: 64 tokens x 64 feats each
      const int row = tid >> 3;          // token within chunk
      const int c0  = (tid & 7) * 8;     // feature offset
      const size_t goff = (size_t)(wstart + c * 64 + row) * DMODEL + c0;
      async_copy_b128(Kg + goff, &Ks[row][c0]);
      uint4 vv = *(const uint4*)(Vg + goff);
      bf16_t tmp[8];
      *(uint4*)tmp = vv;
#pragma unroll
      for (int i = 0; i < 8; ++i) Vt[c0 + i][row] = tmp[i];
    }
#if HAVE_ASYNC_LDS
    WAIT_ASYNC(0);
#endif
    __syncthreads();

    // scores: S = Q * K^T over 64 features (2 k-steps), 4 n-tiles of tokens
    v8f s0 = {}, s1 = {}, s2 = {}, s3 = {};
#pragma unroll
    for (int ks = 0; ks < 2; ++ks) {
      const v16bf a = ks ? aq1 : aq0;
      s0 = wmma_bf16(a, load_b_frag(&Ks[0][ks * 32], 72), s0);
      s1 = wmma_bf16(a, load_b_frag(&Ks[16][ks * 32], 72), s1);
      s2 = wmma_bf16(a, load_b_frag(&Ks[32][ks * 32], 72), s2);
      s3 = wmma_bf16(a, load_b_frag(&Ks[48][ks * 32], 72), s3);
    }

    // online softmax; row data for row (r + 8*lhalf) lives across the 16
    // lanes of this half-wave -> xor-shuffle masks 1,2,4,8 stay in-group.
#pragma unroll
    for (int r = 0; r < 8; ++r) {
      const float v0 = s0[r] * scale, v1 = s1[r] * scale;
      const float v2 = s2[r] * scale, v3 = s3[r] * scale;
      float mc = fmaxf(fmaxf(v0, v1), fmaxf(v2, v3));
#pragma unroll
      for (int off = 8; off >= 1; off >>= 1)
        mc = fmaxf(mc, __shfl_xor(mc, off, 32));
      const float mnew  = fmaxf(rm[r], mc);
      const float alpha = __expf(rm[r] - mnew);
      const float p0 = __expf(v0 - mnew), p1 = __expf(v1 - mnew);
      const float p2 = __expf(v2 - mnew), p3 = __expf(v3 - mnew);
      float rs = p0 + p1 + p2 + p3;
#pragma unroll
      for (int off = 8; off >= 1; off >>= 1) rs += __shfl_xor(rs, off, 32);
      rl[r] = rl[r] * alpha + rs;
      rm[r] = mnew;
      o0[r] *= alpha; o1[r] *= alpha; o2[r] *= alpha; o3[r] *= alpha;

      const int prow = r + 8 * lhalf;   // C layout -> row-major P tile
      Pw[prow * 72 + 0  + lcol] = (bf16_t)p0;
      Pw[prow * 72 + 16 + lcol] = (bf16_t)p1;
      Pw[prow * 72 + 32 + lcol] = (bf16_t)p2;
      Pw[prow * 72 + 48 + lcol] = (bf16_t)p3;
    }

    // O += P * V  (k = 64 chunk tokens -> 2 k-steps; n = 64 head features)
    // Per-wave LDS stream is in-order: P stores precede these loads safely.
#pragma unroll
    for (int ks = 0; ks < 2; ++ks) {
      const v16bf a = load_a_frag(Pw + ks * 32, 72);
      o0 = wmma_bf16(a, load_b_frag(&Vt[0][ks * 32], 72), o0);
      o1 = wmma_bf16(a, load_b_frag(&Vt[16][ks * 32], 72), o1);
      o2 = wmma_bf16(a, load_b_frag(&Vt[32][ks * 32], 72), o2);
      o3 = wmma_bf16(a, load_b_frag(&Vt[48][ks * 32], 72), o3);
    }
  }

  // normalize and store to parity slot (even/odd windows never overlap)
  float* slot = (w & 1) ? slot1 : slot0;
#pragma unroll
  for (int r = 0; r < 8; ++r) {
    const int rowg = wstart + wave * 16 + r + 8 * lhalf;
    const float inv = 1.0f / rl[r];
    const size_t off = ((size_t)b * SEQ + rowg) * DMODEL + h * HDIM + lcol;
    slot[off + 0]  = o0[r] * inv;
    slot[off + 16] = o1[r] * inv;
    slot[off + 32] = o2[r] * inv;
    slot[off + 48] = o3[r] * inv;
  }
}

// ---------------------------------------------------------------------------
extern "C" void kernel_launch(void* const* d_in, const int* in_sizes, int n_in,
                              void* d_out, int out_size, void* d_ws,
                              size_t ws_size, hipStream_t stream) {
  (void)in_sizes; (void)n_in; (void)out_size; (void)ws_size;

  const float* x  = (const float*)d_in[0];
  const float* Wq = (const float*)d_in[1];
  const float* bq = (const float*)d_in[2];
  const float* Wk = (const float*)d_in[3];
  const float* bk = (const float*)d_in[4];
  const float* Wv = (const float*)d_in[5];
  const float* bv = (const float*)d_in[6];
  const float* Wo = (const float*)d_in[7];
  const float* bo = (const float*)d_in[8];
  float* out = (float*)d_out;

  const size_t nx = (size_t)BATCH * SEQ * DMODEL;  // 16,777,216
  const size_t nw = (size_t)DMODEL * DMODEL;       //  1,048,576

  // workspace layout (~310 MB total)
  char* p = (char*)d_ws;
  bf16_t* xb    = (bf16_t*)p; p += nx * 2;
  bf16_t* wqb   = (bf16_t*)p; p += nw * 2;
  bf16_t* wkb   = (bf16_t*)p; p += nw * 2;
  bf16_t* wvb   = (bf16_t*)p; p += nw * 2;
  bf16_t* wob   = (bf16_t*)p; p += nw * 2;
  bf16_t* Qb    = (bf16_t*)p; p += nx * 2;
  bf16_t* Kb    = (bf16_t*)p; p += nx * 2;
  bf16_t* Vb    = (bf16_t*)p; p += nx * 2;
  float*  slot0 = (float*)p;  p += nx * 4;
  float*  slot1 = (float*)p;  p += nx * 4;
  bf16_t* attnb = (bf16_t*)p; p += nx * 2;

  const int M = BATCH * SEQ;   // 16384
  const int N = DMODEL;        // 1024
  const int Kd = DMODEL;       // 1024

  // 1) precision conversion
  convert_f32_to_bf16<<<4096, 256, 0, stream>>>(x, xb, nx);
  convert_f32_to_bf16<<<512, 256, 0, stream>>>(Wq, wqb, nw);
  convert_f32_to_bf16<<<512, 256, 0, stream>>>(Wk, wkb, nw);
  convert_f32_to_bf16<<<512, 256, 0, stream>>>(Wv, wvb, nw);
  convert_f32_to_bf16<<<512, 256, 0, stream>>>(Wo, wob, nw);

  // 2) Q/K/V projections (bf16 out)
  dim3 ggrid(M / 256, N / 64);
  gemm_bias<false><<<ggrid, 256, 0, stream>>>(xb, wqb, bq, Qb, M, N, Kd);
  gemm_bias<false><<<ggrid, 256, 0, stream>>>(xb, wkb, bk, Kb, M, N, Kd);
  gemm_bias<false><<<ggrid, 256, 0, stream>>>(xb, wvb, bv, Vb, M, N, Kd);

  // 3) windowed attention into parity slots
  zero_f32<<<8192, 256, 0, stream>>>(slot0, nx * 2);  // slot0+slot1 contiguous
  attn_kernel<<<dim3(NWIN, NHEADS, BATCH), 512, 0, stream>>>(Qb, Kb, Vb,
                                                             slot0, slot1);

  // 4) average overlapping windows, emit bf16
  combine_slots<<<4096, 256, 0, stream>>>(slot0, slot1, attnb, nx);

  // 5) output projection (fp32 out)
  gemm_bias<true><<<ggrid, 256, 0, stream>>>(attnb, wob, bo, out, M, N, Kd);
}
